// MultiHeadAttention_53429393162920
// MI455X (gfx1250) — matrix operand order repaired and checked by reading, not compiled, not run on hardware
//
#include <hip/hip_runtime.h>
#include <math.h>

typedef _Float16 v16h __attribute__((ext_vector_type(16)));
typedef _Float16 v8h  __attribute__((ext_vector_type(8)));
typedef float    v8f  __attribute__((ext_vector_type(8)));

#define N_HEAD      8
#define D_K         32
#define CONCEPT_NUM 4096
#define MASK_NUM    3000
#define INPUT_DIM   256
#define M_PAD       3008   // 188 * 16 (query rows padded to multiple of 16)

// workspace layout (units: _Float16 elements); every offset is a multiple of 16 (32B aligned)
#define OFF_QF  0
#define OFF_KF  (M_PAD * INPUT_DIM)                   // 770048
#define OFF_WQT (OFF_KF + CONCEPT_NUM * INPUT_DIM)    // 1818624
#define OFF_WKT (OFF_WQT + INPUT_DIM * INPUT_DIM)     // 1884160
#define OFF_QP  (OFF_WKT + INPUT_DIM * INPUT_DIM)     // 1949696
#define OFF_KP  (OFF_QP + M_PAD * INPUT_DIM)          // 2719744
// total = 3768320 halves = ~7.2 MB of d_ws

// ---------------- conversion kernels ----------------

__global__ void k_cvt_pad(const float* __restrict__ src, _Float16* __restrict__ dst,
                          int n_src, int n_dst) {
  int i = blockIdx.x * blockDim.x + threadIdx.x;
  if (i < n_dst) dst[i] = (i < n_src) ? (_Float16)src[i] : (_Float16)0.0f;
}

// wt[n*256 + k] = w[k*256 + n]  (transpose + f32->f16)
__global__ void k_transpose_cvt(const float* __restrict__ w, _Float16* __restrict__ wt) {
  int i = blockIdx.x * blockDim.x + threadIdx.x;   // i = n*256 + k
  int n = i >> 8;
  int k = i & 255;
  wt[i] = (_Float16)w[k * INPUT_DIM + n];
}

// ---------------- projection GEMM: Y[M,256] = X[M,256] @ W (WT row-major) ----------------
// one wave -> one 16x16 tile, K=256 in 8 WMMA steps
__global__ void __launch_bounds__(256) k_proj(const _Float16* __restrict__ X,
                                              const _Float16* __restrict__ WT,
                                              _Float16* __restrict__ Y, int mtiles) {
  int wave = threadIdx.x >> 5;
  int lane = threadIdx.x & 31;
  int tile = blockIdx.x * 8 + wave;
  int mt = tile >> 4;           // 16 tiles along N (=256)
  int nt = tile & 15;
  if (mt >= mtiles) return;     // wave-uniform
  int m  = lane & 15;
  int hi = lane >> 4;

  const _Float16* xrow = X  + (size_t)(mt * 16 + m) * INPUT_DIM;
  const _Float16* wrow = WT + (size_t)(nt * 16 + m) * INPUT_DIM;

  v8f acc = {};
#pragma unroll
  for (int kk = 0; kk < INPUT_DIM; kk += 32) {
    union { v16h v; v8h h[2]; } a;
    a.h[0] = *(const v8h*)(xrow + kk + hi * 8);         // K = hi*8 .. hi*8+7
    a.h[1] = *(const v8h*)(xrow + kk + 16 + hi * 8);    // K = 16+hi*8 ..
    union { v16h v; v8h h[2]; } bu;
    bu.h[0] = *(const v8h*)(wrow + kk + hi * 8);        // K = hi*8 .. hi*8+7
    bu.h[1] = *(const v8h*)(wrow + kk + 16 + hi * 8);   // K = 16+hi*8 ..
    v16h b = bu.v;
    acc = __builtin_amdgcn_wmma_f32_16x16x32_f16(false, a.v, false, b,
                                                 (short)0, acc, false, false);
  }
  int col = nt * 16 + m;
#pragma unroll
  for (int v = 0; v < 8; ++v)
    Y[(size_t)(mt * 16 + v + 8 * hi) * INPUT_DIM + col] = (_Float16)acc[v];
}

// ---------------- zero tail rows [3000,4096) per head ----------------
__global__ void k_zero_tail(float4* __restrict__ out4) {
  long i = (long)blockIdx.x * 256 + threadIdx.x;
  const long per_head = (long)(CONCEPT_NUM - MASK_NUM) * CONCEPT_NUM / 4; // 1122304
  long h   = i / per_head;
  long rem = i - h * per_head;
  float4 z = make_float4(0.f, 0.f, 0.f, 0.f);
  out4[((size_t)h * CONCEPT_NUM + MASK_NUM) * (CONCEPT_NUM / 4) + rem] = z;
}

// ---------------- fused attention + 3-pass softmax + scatter ----------------
// block = 256 threads (8 waves); grid = (188 qblocks, 8 heads)
// wave w owns concept columns [w*512, w*512+512): 32 WMMA tiles per pass.
// Pass 1: row max (fmax only). Pass 2: sum of exp(s - max). Pass 3: write exp*inv_sum.
__global__ void __launch_bounds__(256) k_attn(const _Float16* __restrict__ Qp,
                                              const _Float16* __restrict__ Kp,
                                              const int* __restrict__ qt,
                                              float* __restrict__ out) {
  __shared__ float redm[8][16];
  __shared__ float reds[8][16];
  __shared__ float finm[16];
  __shared__ float fins[16];

  const int qb   = blockIdx.x;
  const int h    = blockIdx.y;
  const int wave = threadIdx.x >> 5;
  const int lane = threadIdx.x & 31;
  const int m    = lane & 15;
  const int hi   = lane >> 4;
  const float scale = 0.17677669529663687f; // 1/sqrt(32)

  // A tile: q rows [qb*16, qb*16+16), head slice h*32..h*32+31 (held for all passes)
  const _Float16* qrow = Qp + (size_t)(qb * 16 + m) * INPUT_DIM + h * D_K;
  union { v16h v; v8h hh[2]; } a;
  a.v = *(const v16h*)(qrow + hi * 16);

  const _Float16* kbase = Kp + (size_t)(wave * 512 + m) * INPUT_DIM + h * D_K + hi * 16;
  const size_t kstride = (size_t)16 * INPUT_DIM;   // 16 concepts per tile

  // ---- pass 1: per-row max over this wave's 512 columns (fmax only) ----
  float rm[8];
#pragma unroll
  for (int v = 0; v < 8; ++v) rm[v] = -INFINITY;

#pragma unroll 2
  for (int t = 0; t < 32; ++t) {
    v16h b = *(const v16h*)(kbase + (size_t)t * kstride);
    v8f acc = {};
    acc = __builtin_amdgcn_wmma_f32_16x16x32_f16(false, a.v, false, b,
                                                 (short)0, acc, false, false);
#pragma unroll
    for (int v = 0; v < 8; ++v) rm[v] = fmaxf(rm[v], acc[v]);
  }
  // max-reduce across the 16 column-lanes inside each half-wave
#pragma unroll
  for (int mask = 1; mask < 16; mask <<= 1) {
#pragma unroll
    for (int v = 0; v < 8; ++v) rm[v] = fmaxf(rm[v], __shfl_xor(rm[v], mask, 32));
  }
  if (m == 0) {
#pragma unroll
    for (int v = 0; v < 8; ++v) redm[wave][v + 8 * hi] = rm[v];
  }
  __syncthreads();
  if (threadIdx.x < 16) {
    float M = redm[0][threadIdx.x];
    for (int w = 1; w < 8; ++w) M = fmaxf(M, redm[w][threadIdx.x]);
    finm[threadIdx.x] = M * scale;     // fold the temperature into the max
  }
  __syncthreads();

  float fm[8];
#pragma unroll
  for (int v = 0; v < 8; ++v) fm[v] = finm[v + 8 * hi];

  // ---- pass 2: sum of exp(s*scale - M) ----
  float rs[8];
#pragma unroll
  for (int v = 0; v < 8; ++v) rs[v] = 0.0f;

#pragma unroll 2
  for (int t = 0; t < 32; ++t) {
    v16h b = *(const v16h*)(kbase + (size_t)t * kstride);
    v8f acc = {};
    acc = __builtin_amdgcn_wmma_f32_16x16x32_f16(false, a.v, false, b,
                                                 (short)0, acc, false, false);
#pragma unroll
    for (int v = 0; v < 8; ++v) rs[v] += __expf(acc[v] * scale - fm[v]);
  }
  // add-reduce across the 16 column-lanes
#pragma unroll
  for (int mask = 1; mask < 16; mask <<= 1) {
#pragma unroll
    for (int v = 0; v < 8; ++v) rs[v] += __shfl_xor(rs[v], mask, 32);
  }
  if (m == 0) {
#pragma unroll
    for (int v = 0; v < 8; ++v) reds[wave][v + 8 * hi] = rs[v];
  }
  __syncthreads();
  if (threadIdx.x < 16) {
    float S = 0.0f;
    for (int w = 0; w < 8; ++w) S += reds[w][threadIdx.x];
    fins[threadIdx.x] = 1.0f / S;
  }
  __syncthreads();

  float fs[8];
  int   orow[8];
#pragma unroll
  for (int v = 0; v < 8; ++v) {
    int r = v + 8 * hi;
    fs[v] = fins[r];
    int grow = qb * 16 + r;
    orow[v] = (grow < MASK_NUM) ? qt[grow] : -1;   // scatter row via qt
  }

  // ---- pass 3: recompute scores, write normalized exp straight to global ----
  for (int t = 0; t < 32; ++t) {
    const int c0 = wave * 512 + t * 16;
    v16h b = *(const v16h*)(kbase + (size_t)t * kstride);
    v8f acc = {};
    acc = __builtin_amdgcn_wmma_f32_16x16x32_f16(false, a.v, false, b,
                                                 (short)0, acc, false, false);
#pragma unroll
    for (int v = 0; v < 8; ++v) {
      if (orow[v] >= 0) {
        out[((size_t)h * CONCEPT_NUM + orow[v]) * CONCEPT_NUM + c0 + m] =
            __expf(acc[v] * scale - fm[v]) * fs[v];
      }
    }
  }
}

// ---------------- host launch ----------------
extern "C" void kernel_launch(void* const* d_in, const int* in_sizes, int n_in,
                              void* d_out, int out_size, void* d_ws, size_t ws_size,
                              hipStream_t stream) {
  const int*   qt    = (const int*)d_in[0];
  const float* query = (const float*)d_in[1];
  const float* key   = (const float*)d_in[2];
  const float* w_q   = (const float*)d_in[3];
  const float* w_k   = (const float*)d_in[4];
  float* out = (float*)d_out;
  _Float16* ws = (_Float16*)d_ws;

  _Float16* qf  = ws + OFF_QF;
  _Float16* kf  = ws + OFF_KF;
  _Float16* wqt = ws + OFF_WQT;
  _Float16* wkt = ws + OFF_WKT;
  _Float16* qp  = ws + OFF_QP;
  _Float16* kp  = ws + OFF_KP;

  // 1) convert inputs to f16 (query zero-padded to 3008 rows)
  k_cvt_pad<<<M_PAD * INPUT_DIM / 256, 256, 0, stream>>>(
      query, qf, MASK_NUM * INPUT_DIM, M_PAD * INPUT_DIM);
  k_cvt_pad<<<CONCEPT_NUM * INPUT_DIM / 256, 256, 0, stream>>>(
      key, kf, CONCEPT_NUM * INPUT_DIM, CONCEPT_NUM * INPUT_DIM);
  k_transpose_cvt<<<INPUT_DIM * INPUT_DIM / 256, 256, 0, stream>>>(w_q, wqt);
  k_transpose_cvt<<<INPUT_DIM * INPUT_DIM / 256, 256, 0, stream>>>(w_k, wkt);

  // 2) projections (WMMA): tiles = (M/16)*(256/16), 8 waves per block
  k_proj<<<(M_PAD / 16) * 16 / 8, 256, 0, stream>>>(qf, wqt, qp, M_PAD / 16);
  k_proj<<<(CONCEPT_NUM / 16) * 16 / 8, 256, 0, stream>>>(kf, wkt, kp, CONCEPT_NUM / 16);

  // 3) zero never-written tail rows [3000,4096) per head
  k_zero_tail<<<(8 * ((CONCEPT_NUM - MASK_NUM) * CONCEPT_NUM / 4)) / 256, 256, 0, stream>>>(
      (float4*)d_out);

  // 4) fused scores + 3-pass softmax + qt-scatter
  k_attn<<<dim3(M_PAD / 16, N_HEAD), 256, 0, stream>>>(qp, kp, qt, out);
}
